// CausalSelfAttention_70884140253706
// MI455X (gfx1250) — matrix operand order repaired and checked by reading, not compiled, not run on hardware
//
#include <hip/hip_runtime.h>
#include <hip/hip_bf16.h>

typedef unsigned short u16;
typedef __attribute__((ext_vector_type(16))) __bf16       v16bf;
typedef __attribute__((ext_vector_type(8)))  __bf16       v8bf;
typedef __attribute__((ext_vector_type(2)))  __bf16       v2bf;
typedef __attribute__((ext_vector_type(8)))  float        v8f;
typedef __attribute__((ext_vector_type(4)))  unsigned int v4u;
typedef __attribute__((ext_vector_type(4)))  int          v4i;

#define N_HEAD   16
#define HEAD_DIM 64
#define CDIM     1024   // N_HEAD * HEAD_DIM
#define SEQ      2048
#define BATCH    4

#if __has_builtin(__builtin_amdgcn_global_load_async_to_lds_b128) && \
    __has_builtin(__builtin_amdgcn_s_wait_asynccnt)
#define ASYNC_LDS 1
typedef __attribute__((address_space(1))) v4i gv4i;   // global int4
typedef __attribute__((address_space(3))) v4i lv4i;   // LDS int4
#else
#define ASYNC_LDS 0
#endif

// 32-byte WMMA operand fragment (16 bf16 per lane)
union Frag {
  v16bf    v;
  v8bf     h[2];
  v4u      q[2];
  unsigned d[8];
  u16      u[16];
};

// f32 -> bf16 via native fptrunc (RNE); backend picks hw cvt if present
__device__ __forceinline__ u16 f2bf(float x) {
  return __builtin_bit_cast(u16, (__bf16)x);
}

// two f32 -> packed bf16x2 in one dword (lo in [15:0], hi in [31:16])
__device__ __forceinline__ unsigned pack2bf(float lo, float hi) {
  v2bf p;
  p[0] = (__bf16)lo;
  p[1] = (__bf16)hi;
  return __builtin_bit_cast(unsigned, p);
}

__device__ __forceinline__ v8f wmma_bf16(const Frag& a, const Frag& b, v8f c) {
  return __builtin_amdgcn_wmma_f32_16x16x32_bf16(
      /*neg_a=*/false, a.v, /*neg_b=*/false, b.v,
      /*c_mod=*/(short)0, c, /*reuse_a=*/false, /*reuse_b=*/false);
}

// ---------------------------------------------------------------------------
// Kernel 1: projection GEMM + fused RoPE, scatter to [B,H,T,D] bf16.
//   mode 0: out = x @ Wq  (all cols RoPE'd)           -> qk_dst
//   mode 1: out = x @ Wkv (cols < C -> K with RoPE -> qk_dst; else V -> v_dst)
// Block: 128 threads (4 waves); tile 64(M) x 64(N); K-step 32.
// ---------------------------------------------------------------------------
__global__ __launch_bounds__(128)
void proj_rope_kernel(const float* __restrict__ x, const float* __restrict__ W,
                      u16* __restrict__ qk_dst, u16* __restrict__ v_dst,
                      int N, int mode)
{
  __shared__ __align__(16) u16 Bt[64][32];   // [n][k] bf16, transposed W tile

  const int tid  = threadIdx.x;
  const int wave = tid >> 5;
  const int lane = tid & 31;
  const int half = lane >> 4;
  const int l16  = lane & 15;

  const int Mb = blockIdx.x * 64;
  const int Nb = blockIdx.y * 64;

  v8f acc[4] = {};

  const float* arow = x + (size_t)(Mb + wave * 16 + l16) * CDIM;

  for (int kk = 0; kk < CDIM; kk += 32) {
    // stage W tile (32 x 64) into LDS: transpose + fp32 -> bf16 (packed dwords)
    {
      const int k2 = (tid & 15) * 2;    // even k: 0..30
      const int nb = (tid >> 4) * 8;    // 0..56
      const float* w0 = W + (size_t)(kk + k2) * N + Nb + nb;
      const float* w1 = w0 + N;
#pragma unroll
      for (int i = 0; i < 8; ++i)
        *(unsigned*)&Bt[nb + i][k2] = pack2bf(w0[i], w1[i]);
      if (kk + 32 < CDIM)
        __builtin_prefetch(w0 + (size_t)32 * N, 0, 1);
    }
    __syncthreads();

    // A fragment direct from global (this code's k order: 16 contiguous K starting at 16*half)
    Frag a;
    a.h[0] = __builtin_convertvector(*(const v8f*)(arow + kk + 16 * half), v8bf);
    a.h[1] = __builtin_convertvector(*(const v8f*)(arow + kk + 16 * half + 8), v8bf);

#pragma unroll
    for (int j = 0; j < 4; ++j) {
      Frag b;   // B in same k order as A: 16 contiguous K starting at 16*half, col n = lane%16
      b.q[0] = *(const v4u*)&Bt[j * 16 + l16][16 * half];
      b.q[1] = *(const v4u*)&Bt[j * 16 + l16][16 * half + 8];
      acc[j] = wmma_bf16(a, b, acc[j]);
    }
    __syncthreads();
  }

  // epilogue: RoPE + scatter into [B,H,T,D] bf16
  const bool rope = (mode == 0) || (Nb < CDIM);
  u16* dst;
  int  cbase;
  if (mode == 0 || Nb < CDIM) { dst = qk_dst; cbase = Nb; }
  else                        { dst = v_dst;  cbase = Nb - CDIM; }
  const int h = cbase >> 6;

#pragma unroll
  for (int r = 0; r < 8; ++r) {
    const int gm = Mb + wave * 16 + r + 8 * half;   // C/D layout: M = r + 8*half
    const int b  = gm >> 11;
    const int t  = gm & (SEQ - 1);
    u16* orow = dst + ((size_t)(b * N_HEAD + h) * SEQ + t) * HEAD_DIM;
    if (rope) {
#pragma unroll
      for (int j = 0; j < 2; ++j) {
        const float x1 = acc[j][r];         // d in [0,32)
        const float x2 = acc[j + 2][r];     // d+32
        const int   d1 = j * 16 + l16;
        const float inv = __expf(-(float)d1 * (9.2103403719761836f / 32.0f));
        float sn, cs;
        __sincosf((float)t * inv, &sn, &cs);
        orow[d1]      = f2bf(x1 * cs - x2 * sn);
        orow[d1 + 32] = f2bf(x2 * cs + x1 * sn);
      }
    } else {
#pragma unroll
      for (int j = 0; j < 4; ++j) orow[j * 16 + l16] = f2bf(acc[j][r]);
    }
  }
}

// ---------------------------------------------------------------------------
// Kernel 2: flash-style causal attention per (b,h,64-query block).
// 4 waves; wave w owns queries [qb*64 + 16w, +16). Key blocks of 32.
// K block staged global->LDS via async-to-LDS (ASYNCcnt) when available.
// ---------------------------------------------------------------------------
__global__ __launch_bounds__(128)
void attn_kernel(const u16* __restrict__ Qws, const u16* __restrict__ Kws,
                 const u16* __restrict__ Vws, u16* __restrict__ Yws)
{
  __shared__ __align__(16) u16 Kl[32][64];      // K block, row-major [key][d]
  __shared__ __align__(16) u16 Vt[64][32];      // [d][key] transposed V block
  __shared__ __align__(16) u16 Pb[4][16][32];   // per-wave P (q x key), C->A relayout

  const int tid  = threadIdx.x;
  const int wave = tid >> 5;
  const int lane = tid & 31;
  const int half = lane >> 4;
  const int l16  = lane & 15;

  const int qb = blockIdx.x;          // 0..31
  const int bh = blockIdx.y;          // b*16 + h
  const int b  = bh >> 4;
  const int h  = bh & 15;

  const u16* Qb = Qws + (size_t)bh * SEQ * HEAD_DIM;
  const u16* Kb = Kws + (size_t)bh * SEQ * HEAD_DIM;
  const u16* Vb = Vws + (size_t)bh * SEQ * HEAD_DIM;

  const int q0 = qb * 64 + wave * 16;

  // Q fragments (two K-steps over D=64), A layout
  Frag qf[2];
  {
    const u16* qrow = Qb + (size_t)(q0 + l16) * HEAD_DIM;
#pragma unroll
    for (int s = 0; s < 2; ++s) {
      qf[s].q[0] = *(const v4u*)(qrow + s * 32 + 16 * half);
      qf[s].q[1] = *(const v4u*)(qrow + s * 32 + 16 * half + 8);
    }
  }

  float mrow[8], lrow[8];
#pragma unroll
  for (int r = 0; r < 8; ++r) { mrow[r] = -1e30f; lrow[r] = 0.0f; }
  v8f o[4] = {};

  const int nkb = qb * 2 + 2;          // causal: keys up to end of query block
  for (int kb = 0; kb < nkb; ++kb) {
    const int kbase = kb * 32;

    // stage K block (contiguous 4KB) into LDS
    {
      const u16* gsrc = Kb + (size_t)kbase * HEAD_DIM;
#pragma unroll
      for (int c = 0; c < 2; ++c) {
        const int off = (tid * 2 + c) * 8;   // 16B chunks, 128 thr * 32B = 4KB
#if ASYNC_LDS
        __builtin_amdgcn_global_load_async_to_lds_b128(
            (gv4i*)(gsrc + off), (lv4i*)(&Kl[0][0] + off), 0, 0);
#else
        *(v4u*)(&Kl[0][0] + off) = *(const v4u*)(gsrc + off);
#endif
      }
    }

    // stage V block (32 keys x 64 d) transposed into LDS
    {
      const int key = tid >> 2;
      const int db  = (tid & 3) * 16;
      const u16* vrow = Vb + (size_t)(kbase + key) * HEAD_DIM + db;
#pragma unroll
      for (int i = 0; i < 16; ++i) Vt[db + i][key] = vrow[i];
    }
#if ASYNC_LDS
    __builtin_amdgcn_s_wait_asynccnt(0);
#endif
    __syncthreads();

    // S = Q @ K^T  (B operand: K rows from LDS, contiguous in d)
    v8f sfr[2];
#pragma unroll
    for (int ks = 0; ks < 2; ++ks) {
      v8f s = {};
      const u16* krow = &Kl[ks * 16 + l16][0];
#pragma unroll
      for (int dstep = 0; dstep < 2; ++dstep) {
        Frag kf;
        kf.q[0] = *(const v4u*)(krow + dstep * 32 + 16 * half);
        kf.q[1] = *(const v4u*)(krow + dstep * 32 + 16 * half + 8);
        s = wmma_bf16(qf[dstep], kf, s);
      }
      sfr[ks] = s;
    }

    // scale + causal mask
    float sc[2][8];
#pragma unroll
    for (int ks = 0; ks < 2; ++ks) {
      const int key = kbase + ks * 16 + l16;
#pragma unroll
      for (int r = 0; r < 8; ++r) {
        const int   qr = q0 + r + 8 * half;
        const float v  = sfr[ks][r] * 0.125f;   // 1/sqrt(64)
        sc[ks][r] = (key <= qr) ? v : -3.0e38f;
      }
    }

    // online softmax; rows live in 16-lane groups (C layout)
#pragma unroll
    for (int r = 0; r < 8; ++r) {
      float m = fmaxf(sc[0][r], sc[1][r]);
#pragma unroll
      for (int off = 1; off < 16; off <<= 1)
        m = fmaxf(m, __shfl_xor(m, off, 32));
      const float mnew  = fmaxf(mrow[r], m);
      const float alpha = __expf(mrow[r] - mnew);
      const float p0 = __expf(sc[0][r] - mnew);
      const float p1 = __expf(sc[1][r] - mnew);
      float ps = p0 + p1;
#pragma unroll
      for (int off = 1; off < 16; off <<= 1)
        ps += __shfl_xor(ps, off, 32);
      lrow[r] = lrow[r] * alpha + ps;
      mrow[r] = mnew;
#pragma unroll
      for (int j = 0; j < 4; ++j) o[j][r] *= alpha;
      Pb[wave][r + 8 * half][l16]      = f2bf(p0);
      Pb[wave][r + 8 * half][16 + l16] = f2bf(p1);
    }
    __syncthreads();

    // O += P @ V
    Frag pf;
    {
      const u16* prow = &Pb[wave][l16][0];
      pf.q[0] = *(const v4u*)(prow + 16 * half);
      pf.q[1] = *(const v4u*)(prow + 16 * half + 8);
    }
#pragma unroll
    for (int j = 0; j < 4; ++j) {
      Frag vf;
      vf.q[0] = *(const v4u*)&Vt[j * 16 + l16][16 * half];
      vf.q[1] = *(const v4u*)&Vt[j * 16 + l16][16 * half + 8];
      o[j] = wmma_bf16(pf, vf, o[j]);
    }
    __syncthreads();
  }

  // normalize + store y as [B,T,C] bf16
#pragma unroll
  for (int r = 0; r < 8; ++r) {
    const float invl = 1.0f / lrow[r];
    const int   t    = q0 + r + 8 * half;
    u16* yrow = Yws + (size_t)(b * SEQ + t) * CDIM + h * HEAD_DIM;
#pragma unroll
    for (int j = 0; j < 4; ++j)
      yrow[j * 16 + l16] = f2bf(o[j][r] * invl);
  }
}

// ---------------------------------------------------------------------------
// Kernel 3: out = y @ Wc, fp32 output.
// ---------------------------------------------------------------------------
__global__ __launch_bounds__(128)
void out_proj_kernel(const u16* __restrict__ Y, const float* __restrict__ W,
                     float* __restrict__ out)
{
  __shared__ __align__(16) u16 Bt[64][32];

  const int tid  = threadIdx.x;
  const int wave = tid >> 5;
  const int lane = tid & 31;
  const int half = lane >> 4;
  const int l16  = lane & 15;

  const int Mb = blockIdx.x * 64;
  const int Nb = blockIdx.y * 64;

  v8f acc[4] = {};
  const u16* arow = Y + (size_t)(Mb + wave * 16 + l16) * CDIM;

  for (int kk = 0; kk < CDIM; kk += 32) {
    {
      const int k2 = (tid & 15) * 2;
      const int nb = (tid >> 4) * 8;
      const float* w0 = W + (size_t)(kk + k2) * CDIM + Nb + nb;
      const float* w1 = w0 + CDIM;
#pragma unroll
      for (int i = 0; i < 8; ++i)
        *(unsigned*)&Bt[nb + i][k2] = pack2bf(w0[i], w1[i]);
      if (kk + 32 < CDIM)
        __builtin_prefetch(w0 + (size_t)32 * CDIM, 0, 1);
    }
    __syncthreads();

    Frag a;   // A rows already bf16: two 16B chunks
    a.q[0] = *(const v4u*)(arow + kk + 16 * half);
    a.q[1] = *(const v4u*)(arow + kk + 16 * half + 8);

#pragma unroll
    for (int j = 0; j < 4; ++j) {
      Frag b;
      b.q[0] = *(const v4u*)&Bt[j * 16 + l16][16 * half];
      b.q[1] = *(const v4u*)&Bt[j * 16 + l16][16 * half + 8];
      acc[j] = wmma_bf16(a, b, acc[j]);
    }
    __syncthreads();
  }

#pragma unroll
  for (int r = 0; r < 8; ++r) {
    const int gm = Mb + wave * 16 + r + 8 * half;
    float* orow = out + (size_t)gm * CDIM + Nb;
#pragma unroll
    for (int j = 0; j < 4; ++j) orow[j * 16 + l16] = acc[j][r];
  }
}

// ---------------------------------------------------------------------------
extern "C" void kernel_launch(void* const* d_in, const int* in_sizes, int n_in,
                              void* d_out, int out_size, void* d_ws, size_t ws_size,
                              hipStream_t stream)
{
  (void)in_sizes; (void)n_in; (void)out_size; (void)ws_size;
  const float* x   = (const float*)d_in[0];
  const float* Wq  = (const float*)d_in[1];
  const float* Wkv = (const float*)d_in[2];
  const float* Wc  = (const float*)d_in[3];
  float* out = (float*)d_out;

  const size_t ELEMS = (size_t)BATCH * N_HEAD * SEQ * HEAD_DIM;  // 8M
  u16* Qws = (u16*)d_ws;
  u16* Kws = Qws + ELEMS;
  u16* Vws = Kws + ELEMS;
  u16* Yws = Vws + ELEMS;

  dim3 blk(128);
  // Q = rope(x @ Wq) -> [B,H,T,D]
  proj_rope_kernel<<<dim3(128, 16), blk, 0, stream>>>(x, Wq,  Qws, nullptr, CDIM,     0);
  // K = rope(x @ Wkv[:, :C]), V = x @ Wkv[:, C:]
  proj_rope_kernel<<<dim3(128, 32), blk, 0, stream>>>(x, Wkv, Kws, Vws,     2 * CDIM, 1);
  // causal flash attention -> y [B,T,C] bf16
  attn_kernel<<<dim3(32, 64), blk, 0, stream>>>(Qws, Kws, Vws, Yws);
  // out = y @ Wc (fp32)
  out_proj_kernel<<<dim3(128, 16), blk, 0, stream>>>(Yws, Wc, out);
}
